// MemoryBank2D_82025285419168
// MI455X (gfx1250) — compile-verified, matrix operand order checked by reading, not run on hardware
//
#include <hip/hip_runtime.h>
#include <hip/hip_bf16.h>
#include <math.h>

// MemoryBank2D fused kernel for MI455X (gfx1250, wave32, WMMA + TDM).
//
// x:      [16, 512, 64, 64] f32   (B,C,H,W; w fastest)
// memory: [1024, 512] f32
// out:    recon [16,512,64,64] f32  ++  match_map [16,64,64] f32
//
// Roofline: 137 GFLOP matrix work vs ~270MB HBM -> compute-bound at f16 WMMA
// rates; both GEMMs run on v_wmma_f32_16x16x32_f16 with f32 accumulation.
// The 64KB per-WG x-tile (32 w x 512 c, row stride H*W) is staged into LDS
// by the Tensor Data Mover (one tensor_load_to_lds per WG) instead of a
// VGPR round-trip, freeing VALU/VMEM issue slots for the WMMA pipeline.

typedef _Float16 half8  __attribute__((ext_vector_type(8)));
typedef _Float16 half16 __attribute__((ext_vector_type(16)));
typedef float    f32x8  __attribute__((ext_vector_type(8)));
typedef unsigned int v4u __attribute__((ext_vector_type(4)));
typedef int          v4i __attribute__((ext_vector_type(4)));
typedef int          v8i __attribute__((ext_vector_type(8)));

#define C_DIM   512
#define K_DIM   1024
#define MT      32          // pixels per workgroup
#define A_STR   536         // halves per row of A tile in LDS (pad for banks)
#define S_STR   1032        // halves per row of sim/attn tile in LDS
#define HW      4096        // H*W
#define EPS     1e-12f

#if __has_builtin(__builtin_amdgcn_tensor_load_to_lds)
#define HAVE_TDM 1
#endif

// Load one 16-bit WMMA fragment (16 halves) for this lane.
// base must already include the lane's row offset and the lane-group select
// (+8 halves for lanes 16..31); layout per CDNA5 ISA 16-bit A matrix:
//   lanes 0-15 : K = {off..off+7, off+16..off+23}
//   lanes 16-31: K = {off+8..off+15, off+24..off+31}
static __device__ inline half16 ld_frag(const _Float16* __restrict__ base, int off) {
    half8 a = *(const half8*)(base + off);
    half8 b = *(const half8*)(base + off + 16);
    return __builtin_shufflevector(a, b, 0,1,2,3,4,5,6,7,8,9,10,11,12,13,14,15);
}

// ---------------------------------------------------------------------------
// Kernel 0: L2-normalize memory rows; emit f16 row-major (memF) and
// f16 transposed (memT) copies into workspace. 1024 blocks x 256 threads.
// ---------------------------------------------------------------------------
__global__ __launch_bounds__(256)
void mb_norm_mem(const float* __restrict__ mem,
                 _Float16* __restrict__ memF,
                 _Float16* __restrict__ memT) {
    __shared__ float red[256];
    __shared__ float s_rn;
    const int r = blockIdx.x;
    const int t = threadIdx.x;

    float v0 = mem[(size_t)r * C_DIM + t];
    float v1 = mem[(size_t)r * C_DIM + 256 + t];
    red[t] = v0 * v0 + v1 * v1;
    __syncthreads();
    for (int off = 128; off > 0; off >>= 1) {
        if (t < off) red[t] += red[t + off];
        __syncthreads();
    }
    if (t == 0) s_rn = 1.0f / fmaxf(sqrtf(red[0]), EPS);
    __syncthreads();
    const float rn = s_rn;

    _Float16 h0 = (_Float16)(v0 * rn);
    _Float16 h1 = (_Float16)(v1 * rn);
    memF[(size_t)r * C_DIM + t]         = h0;
    memF[(size_t)r * C_DIM + 256 + t]   = h1;
    memT[(size_t)t * K_DIM + r]         = h0;   // memT[c][k]
    memT[(size_t)(t + 256) * K_DIM + r] = h1;
}

// ---------------------------------------------------------------------------
// Main fused kernel: 2048 blocks x 256 threads (8 waves).
// Each block: 32 pixels (one b, one h, half of w).
// ---------------------------------------------------------------------------
__global__ __launch_bounds__(256)
void mb_main(const float* __restrict__ x,
             const _Float16* __restrict__ memF,
             const _Float16* __restrict__ memT,
             float* __restrict__ recon,
             float* __restrict__ match) {
    // region1: phase A = x staging f32 [512][32] (64KB, TDM destination)
    //          phase B/C = sim/attn f16 [32][S_STR] (64.5KB)
    __shared__ __align__(16) unsigned char region1[MT * S_STR * 2];
    __shared__ __align__(16) _Float16 sA[MT * A_STR];     // feat_norm f16 tile
    __shared__ float ssqp[8 * MT];
    __shared__ float rnorm[MT];
    __shared__ float stm[MT * 8];
    __shared__ float sts[MT * 8];
    __shared__ float rowm[MT];
    __shared__ float rowrs[MT];

    float*    xs   = (float*)region1;
    _Float16* simH = (_Float16*)region1;

    const int t    = threadIdx.x;
    const int lane = t & 31;
    const int wv   = t >> 5;
    const int wg   = blockIdx.x;

    // wg -> (b, h, w0): 128 WGs per batch image, 2 per h-row
    const int b   = wg >> 7;
    const int rem = wg & 127;
    const int h   = rem >> 1;
    const int w0  = (rem & 1) * MT;
    const size_t xbase = (size_t)b * (C_DIM * HW) + (size_t)h * 64 + w0;

    // ---------------- Phase A: stage x tile -> LDS, L2-normalize -----------
    const int p  = t & 31;        // pixel (w offset)
    const int cg = wv;            // channel group: 64 channels per wave
#ifdef HAVE_TDM
    // One TDM descriptor per WG: 2D tile, 32 (w, contiguous) x 512 (c) f32,
    // global row stride = H*W elements, destination = xs[c][p] row-major.
    if (wv == 0) {
        const unsigned long long ga = (unsigned long long)(uintptr_t)(x + xbase);
        const unsigned int ldsa = (unsigned int)(uintptr_t)xs;
        v4u g0;
        g0[0] = 1u;                                          // count=1 valid D#
        g0[1] = ldsa;                                        // lds_addr
        g0[2] = (unsigned int)(ga & 0xFFFFFFFFu);            // global_addr[31:0]
        g0[3] = (unsigned int)((ga >> 32) & 0x01FFFFFFu)     // global_addr[56:32]
              | (2u << 30);                                  // type = 2 (image)
        v8i g1;
        g1[0] = (2 << 16);        // workgroup_mask=0, data_size=2 (4 bytes)
        g1[1] = (32 << 16);       // tensor_dim0 = 32  (bits[79:48] low half)
        g1[2] = (512 << 16);      // tensor_dim0 hi=0, tensor_dim1 = 512 (low half)
        g1[3] = (32 << 16);       // tensor_dim1 hi=0, tile_dim0 = 32
        g1[4] = 512;              // tile_dim1 = 512, tile_dim2 = 0
        g1[5] = 4096;             // tensor_dim0_stride = H*W (bits[207:160] low)
        g1[6] = 0;                // stride0 hi, tensor_dim1_stride low
        g1[7] = 0;
        v4i z4 = {0, 0, 0, 0};
#if __clang_major__ >= 23
        v8i z8 = {0, 0, 0, 0, 0, 0, 0, 0};
        __builtin_amdgcn_tensor_load_to_lds(g0, g1, z4, z4, z8, 0);
#else
        __builtin_amdgcn_tensor_load_to_lds(g0, g1, z4, z4, 0);
#endif
        __builtin_amdgcn_s_wait_tensorcnt(0);
    }
    __syncthreads();              // x tile resident in LDS for all waves
    {
        float ssq = 0.0f;
        #pragma unroll 4
        for (int i = 0; i < 64; ++i) {
            const int c = cg * 64 + i;
            const float v = xs[c * MT + p];
            ssq += v * v;
        }
        ssqp[cg * MT + p] = ssq;
    }
    __syncthreads();
#else
    {
        float ssq = 0.0f;
        #pragma unroll 4
        for (int i = 0; i < 64; ++i) {
            const int c = cg * 64 + i;
            float v = __builtin_nontemporal_load(&x[xbase + (size_t)c * HW + p]);
            xs[c * MT + p] = v;
            ssq += v * v;
        }
        ssqp[cg * MT + p] = ssq;
    }
    __syncthreads();
#endif
    if (t < MT) {
        float s = 0.0f;
        #pragma unroll
        for (int j = 0; j < 8; ++j) s += ssqp[j * MT + t];
        rnorm[t] = 1.0f / fmaxf(sqrtf(s), EPS);
    }
    __syncthreads();
    {
        const float rn = rnorm[p];
        #pragma unroll 4
        for (int i = 0; i < 64; ++i) {
            const int c = cg * 64 + i;
            sA[p * A_STR + c] = (_Float16)(xs[c * MT + p] * rn);
        }
    }
    __syncthreads();   // sA ready; region1 free for sim

    const int rb  = wv & 1;             // row-block (16 pixels) owned by wave
    const int q   = wv >> 1;            // quarter of the N / C space
    const int mr  = rb * 16;
    const int l15 = lane & 15;
    const int sel = (lane & 16) ? 8 : 0;
    const int hi  = lane >> 4;

    // ---------------- Phase B: sim = feat_norm @ mem_normT  (K=512) --------
    {
        const _Float16* aBase = &sA[(mr + l15) * A_STR + sel];
        for (int j = 0; j < 16; ++j) {                    // 16 n-blocks of 16
            const int nb = q * 256 + j * 16;
            const _Float16* bBase = memF + (size_t)(nb + l15) * C_DIM + sel;
            f32x8 acc = {0.f,0.f,0.f,0.f,0.f,0.f,0.f,0.f};
            #pragma unroll
            for (int kc = 0; kc < 16; ++kc) {             // K = 16 * 32 = 512
                half16 af = ld_frag(aBase, kc * 32);
                half16 bf = ld_frag(bBase, kc * 32);
                acc = __builtin_amdgcn_wmma_f32_16x16x32_f16(
                          false, af, false, bf, (short)0, acc, false, false);
            }
            #pragma unroll
            for (int v = 0; v < 8; ++v)                   // D: m = v + 8*hi, n = l15
                simH[(mr + v + 8 * hi) * S_STR + nb + l15] = (_Float16)acc[v];
        }
    }
    __syncthreads();

    // ---------------- Phase C0: softmax stats (max, sum-exp) per pixel -----
    {
        const int pr = t >> 3;          // pixel row 0..31
        const int sg = t & 7;           // segment of 128 n-values
        const _Float16* srow = simH + pr * S_STR + sg * 128;
        float m = -1e30f;
        #pragma unroll 8
        for (int i = 0; i < 128; ++i) m = fmaxf(m, (float)srow[i]);
        float s = 0.0f;
        #pragma unroll 8
        for (int i = 0; i < 128; ++i) s += __expf((float)srow[i] - m);
        stm[pr * 8 + sg] = m;
        sts[pr * 8 + sg] = s;
    }
    __syncthreads();
    if (t < MT) {
        float m = -1e30f;
        #pragma unroll
        for (int j = 0; j < 8; ++j) m = fmaxf(m, stm[t * 8 + j]);
        float s = 0.0f;
        #pragma unroll
        for (int j = 0; j < 8; ++j) s += sts[t * 8 + j] * __expf(stm[t * 8 + j] - m);
        const float rs = 1.0f / s;
        rowm[t]  = m;
        rowrs[t] = rs;
        match[(size_t)wg * MT + t] = rs;     // max attn = exp(max-m)/s = 1/s
    }
    __syncthreads();

    // ---------------- Phase C1: attn = exp(sim - m) / s, in place ----------
    {
        const int pr = t >> 3;
        const int sg = t & 7;
        const float m  = rowm[pr];
        const float rs = rowrs[pr];
        _Float16* arow = simH + pr * S_STR + sg * 128;
        #pragma unroll 8
        for (int i = 0; i < 128; ++i)
            arow[i] = (_Float16)(__expf((float)arow[i] - m) * rs);
    }
    __syncthreads();

    // ---------------- Phase C2: recon = attn @ mem_norm  (K=1024) ----------
    {
        const int cq = q * 128;                          // 128 channels per wave
        const _Float16* aBase = simH + (mr + l15) * S_STR + sel;
        for (int j = 0; j < 8; ++j) {                    // 8 c-blocks of 16
            const int cb = cq + j * 16;
            const _Float16* bBase = memT + (size_t)(cb + l15) * K_DIM + sel;
            f32x8 acc = {0.f,0.f,0.f,0.f,0.f,0.f,0.f,0.f};
            #pragma unroll 8
            for (int kc = 0; kc < 32; ++kc) {            // K = 32 * 32 = 1024
                half16 af = ld_frag(aBase, kc * 32);
                half16 bf = ld_frag(bBase, kc * 32);
                acc = __builtin_amdgcn_wmma_f32_16x16x32_f16(
                          false, af, false, bf, (short)0, acc, false, false);
            }
            #pragma unroll
            for (int v = 0; v < 8; ++v) {
                const int pp = mr + v + 8 * hi;          // pixel (w) index
                const int cc = cb + l15;                 // channel
                __builtin_nontemporal_store(
                    acc[v], &recon[xbase + (size_t)cc * HW + pp]);
            }
        }
    }
}

// ---------------------------------------------------------------------------
extern "C" void kernel_launch(void* const* d_in, const int* in_sizes, int n_in,
                              void* d_out, int out_size, void* d_ws, size_t ws_size,
                              hipStream_t stream) {
    const float* x   = (const float*)d_in[0];   // [16,512,64,64]
    const float* mem = (const float*)d_in[1];   // [1024,512]

    _Float16* memF = (_Float16*)d_ws;                    // [1024][512] f16
    _Float16* memT = memF + (size_t)K_DIM * C_DIM;       // [512][1024] f16

    float* recon = (float*)d_out;                        // [16,512,64,64]
    float* match = recon + (size_t)in_sizes[0];          // [16,64,64]

    mb_norm_mem<<<K_DIM, 256, 0, stream>>>(mem, memF, memT);

    const int n_pixels = in_sizes[0] / C_DIM;            // 65536
    mb_main<<<n_pixels / MT, 256, 0, stream>>>(x, memF, memT, recon, match);
}
